// Attention_5660766896599
// MI455X (gfx1250) — matrix operand order repaired and checked by reading, not compiled, not run on hardware
//
#include <hip/hip_runtime.h>

// ---------------------------------------------------------------- constants
constexpr int B_  = 8;
constexpr int TQ_ = 2048;
constexpr int TK_ = 2048;
constexpr int DD  = 1024;

// ---------------------------------------------------------------- types
typedef __attribute__((ext_vector_type(16))) __bf16          v16bf;
typedef __attribute__((ext_vector_type(8)))  float           v8f;
typedef __attribute__((ext_vector_type(8)))  unsigned short  ushort8;
typedef __attribute__((ext_vector_type(2)))  unsigned int    uint2v;
typedef __attribute__((ext_vector_type(4)))  unsigned int    u32x4;
typedef __attribute__((ext_vector_type(4)))  int             i32x4;
typedef __attribute__((ext_vector_type(8)))  int             i32x8;

union FragU { v16bf v; ushort8 h[2]; };

__device__ __forceinline__ v16bf ld_frag(const unsigned short* p0,
                                         const unsigned short* p1) {
  FragU u;
  u.h[0] = *(const ushort8*)p0;
  u.h[1] = *(const ushort8*)p1;
  return u.v;
}

__device__ __forceinline__ v8f wmma_bf16(v16bf a, v16bf b, v8f c) {
  return __builtin_amdgcn_wmma_f32_16x16x32_bf16(
      false, a, false, b, (short)0, c, false, false);
}

// packed f32x2 -> bf16x2 (RNE) in one VALU op
__device__ __forceinline__ unsigned cvt_pk_bf16(float lo, float hi) {
  unsigned r;
  asm("v_cvt_pk_bf16_f32 %0, %1, %2" : "=v"(r) : "v"(lo), "v"(hi));
  return r;
}

__device__ __forceinline__ unsigned lds_off(const void* p) {
  return (unsigned)(unsigned long long)(uintptr_t)p;   // low 32b = LDS offset
}

// --------------------------------------------------------------- TDM helper
// 2D tile load: tile_w x tile_h bf16 elems, row stride in elems, optional
// LDS padding: pad_amount_code+1 dwords every 2^(pad_interval_code+1) dwords.
__device__ __forceinline__ void tdm_load_2d(unsigned lds_byte, const void* gp,
                                            unsigned tile_w, unsigned tile_h,
                                            unsigned row_stride,
                                            unsigned pad_icode,
                                            unsigned pad_acode, bool pad_en) {
  unsigned long long ga = (unsigned long long)(uintptr_t)gp;
  u32x4 g0;
  g0[0] = 1u;                                            // count=1 (valid D#)
  g0[1] = lds_byte;                                      // lds_addr
  g0[2] = (unsigned)ga;                                  // global_addr lo
  g0[3] = (unsigned)((ga >> 32) & 0x01ffffffu) | 0x80000000u;  // hi | type=2
  unsigned w0 = (1u << 16);                              // data_size = 2B
  if (pad_en) w0 |= (1u << 20) | (pad_icode << 22) | (pad_acode << 25);
  i32x8 g1 = {
      (int)w0,
      (int)(tile_w << 16),                               // tensor_dim0 lo16
      (int)((tile_w >> 16) | (tile_h << 16)),            // dim0 hi | dim1 lo
      (int)(tile_w << 16),                               // dim1 hi=0 | tile_dim0
      (int)tile_h,                                       // tile_dim1 | tile_dim2=0
      (int)row_stride,                                   // dim0_stride lo32
      0, 0};
  i32x4 z4 = {0, 0, 0, 0};
#if defined(__clang_major__) && __clang_major__ >= 23
  i32x8 z8 = {0, 0, 0, 0, 0, 0, 0, 0};
  __builtin_amdgcn_tensor_load_to_lds(g0, g1, z4, z4, z8, 0);
#else
  __builtin_amdgcn_tensor_load_to_lds(g0, g1, z4, z4, 0);
#endif
}

// ================================================================ GEMM:
//   out[M,1024](bf16) = X[M,1024](f32) @ W[1024,1024](f32) + bias
__global__ __launch_bounds__(256) void proj_gemm(
    const float* __restrict__ X, const float* __restrict__ W,
    const float* __restrict__ bias, unsigned short* __restrict__ out) {
  __shared__ unsigned short As[128 * 48];   // [row][k], stride 48 halfs (96B)
  __shared__ unsigned short Bs[128 * 48];   // [n][k] (W transposed)

  const int tid  = threadIdx.x;
  const int lane = tid & 31;
  const int wave = tid >> 5;
  const int wm   = wave >> 1;
  const int wn   = wave & 1;
  const size_t m0 = (size_t)blockIdx.x * 128;
  const int    n0 = blockIdx.y * 128;
  const int ln16 = lane & 15;
  const int hi   = lane >> 4;
  const int akb  = hi ? 8 : 0;
  const int bkb  = hi ? 8 : 0;

  v8f zero = {};
  v8f acc[2][4];
  for (int mi = 0; mi < 2; ++mi)
    for (int ni = 0; ni < 4; ++ni) acc[mi][ni] = zero;

#pragma unroll 1
  for (int kt = 0; kt < DD / 32; ++kt) {
    __syncthreads();
    // stage A: 128 rows x 32 k (fp32 -> packed bf16)
    for (int i = 0; i < 4; ++i) {
      int lin = i * 256 + tid;
      int r = lin >> 3, c4 = lin & 7;
      float4 f = *(const float4*)&X[(m0 + r) * DD + kt * 32 + c4 * 4];
      uint2v p;
      p[0] = cvt_pk_bf16(f.x, f.y);
      p[1] = cvt_pk_bf16(f.z, f.w);
      *(uint2v*)&As[r * 48 + c4 * 4] = p;
    }
    // stage W transposed: 32 k-rows x 128 n -> Bs[n][k]
    for (int i = 0; i < 4; ++i) {
      int lin = i * 256 + tid;
      int k = lin >> 5, c4 = lin & 31;
      float4 f = *(const float4*)&W[(size_t)(kt * 32 + k) * DD + n0 + c4 * 4];
      int nn = c4 * 4;
      unsigned p01 = cvt_pk_bf16(f.x, f.y);
      unsigned p23 = cvt_pk_bf16(f.z, f.w);
      Bs[(nn + 0) * 48 + k] = (unsigned short)p01;
      Bs[(nn + 1) * 48 + k] = (unsigned short)(p01 >> 16);
      Bs[(nn + 2) * 48 + k] = (unsigned short)p23;
      Bs[(nn + 3) * 48 + k] = (unsigned short)(p23 >> 16);
    }
    __syncthreads();

    v16bf a[2], bf[4];
    for (int mi = 0; mi < 2; ++mi) {
      const unsigned short* p = &As[(wm * 32 + mi * 16 + ln16) * 48 + akb];
      a[mi] = ld_frag(p, p + 16);
    }
    for (int ni = 0; ni < 4; ++ni) {
      const unsigned short* p = &Bs[(wn * 64 + ni * 16 + ln16) * 48 + bkb];
      bf[ni] = ld_frag(p, p + 16);
    }
    for (int mi = 0; mi < 2; ++mi)
      for (int ni = 0; ni < 4; ++ni)
        acc[mi][ni] = wmma_bf16(a[mi], bf[ni], acc[mi][ni]);
  }

  for (int ni = 0; ni < 4; ++ni) {
    int col = n0 + wn * 64 + ni * 16 + ln16;
    float bv = bias[col];
    for (int mi = 0; mi < 2; ++mi)
      for (int v = 0; v < 8; ++v) {
        size_t row = m0 + wm * 32 + mi * 16 + v + 8 * hi;
        float x = acc[mi][ni][v] + bv;
        out[row * DD + col] = (unsigned short)cvt_pk_bf16(x, x);
      }
  }
}

// ================================================================ Flash attn
// block: 128 thr = 4 waves (wq: 16 q-rows; wd: 256 d-cols)
// grid: (TQ/32, D/512, B). Q and K tiles staged by the Tensor Data Mover.
__global__ __launch_bounds__(128) void flash_attn(
    const unsigned short* __restrict__ qb, const unsigned short* __restrict__ kb,
    const unsigned short* __restrict__ vb, const int* __restrict__ mask,
    float* __restrict__ out) {
  __shared__ unsigned short Qs[32 * 1040];   // TDM pads 8 halfs per 512
  __shared__ unsigned short Ks[32 * 520];    // 512-half rows + 8 pad
  __shared__ unsigned short Vt[512 * 40];    // [d_local][k] transposed
  __shared__ unsigned short Ps[4 * 16 * 40]; // per-wave P transpose scratch

  const int tid  = threadIdx.x;
  const int lane = tid & 31;
  const int wave = tid >> 5;
  const int wq   = wave >> 1;
  const int wd   = wave & 1;
  const int q0   = blockIdx.x * 32;
  const int d0   = blockIdx.y * 512;
  const int b    = blockIdx.z;
  const int ln16 = lane & 15;
  const int hi   = lane >> 4;
  const int akb  = hi ? 8 : 0;
  const int bkb  = hi ? 8 : 0;
  const float scale = 0.03125f;              // 1/sqrt(1024)

  // --- stage Q once via TDM (32 x 1024 bf16, padded rows -> stride 1040)
  if (wave == 0) {
    tdm_load_2d(lds_off(Qs), &qb[((size_t)b * TQ_ + q0) * DD],
                1024, 32, 1024, /*interval 256dw*/ 7, /*amount 4dw*/ 3, true);
    __builtin_amdgcn_s_wait_tensorcnt((short)0);
  }
  __syncthreads();

  v8f zero = {};
  v8f O[16];
  for (int ni = 0; ni < 16; ++ni) O[ni] = zero;
  float m_i[8], l_i[8];
  for (int v = 0; v < 8; ++v) { m_i[v] = -1e30f; l_i[v] = 0.0f; }

  const int qrow = wq * 16 + ln16;
  const size_t qg = (size_t)b * TQ_ + q0 + wq * 16;

#pragma unroll 1
  for (int kt = 0; kt < TK_ / 32; ++kt) {
    // prefetch next tile's mask rows into L2 (global_prefetch_b8)
    if (kt + 1 < TK_ / 32)
      __builtin_prefetch(
          &mask[(qg + ln16) * (size_t)TK_ + (kt + 1) * 32 + hi * 16], 0, 1);

    v8f s0 = zero, s1 = zero;
#pragma unroll 1
    for (int dh = 0; dh < 2; ++dh) {
      __syncthreads();
      // --- stage K half-tile via TDM (32 x 512, row stride 1024, padded)
      if (wave == 0) {
        tdm_load_2d(lds_off(Ks),
                    &kb[((size_t)b * TK_ + kt * 32) * DD + dh * 512],
                    512, 32, 1024, 7, 3, true);
        __builtin_amdgcn_s_wait_tensorcnt((short)0);
      }
      if (dh == 0) {
        // stage V tile transposed: pack 2 k-rows/b32 with v_perm_b32
        for (int i = 0; i < 8; ++i) {
          int lin = i * 128 + tid;             // 16 k-pairs x 64 d8
          int kp = lin >> 6, d8 = lin & 63;
          const unsigned short* src =
              &vb[((size_t)b * TK_ + kt * 32 + kp * 2) * DD + d0 + d8 * 8];
          uint4 r0 = *(const uint4*)src;
          uint4 r1 = *(const uint4*)(src + DD);
          unsigned a0[4] = {r0.x, r0.y, r0.z, r0.w};
          unsigned a1[4] = {r1.x, r1.y, r1.z, r1.w};
          for (int j = 0; j < 4; ++j) {
            unsigned lo = __builtin_amdgcn_perm(a1[j], a0[j], 0x05040100u);
            unsigned hh = __builtin_amdgcn_perm(a1[j], a0[j], 0x07060302u);
            *(unsigned*)&Vt[(d8 * 8 + 2 * j) * 40 + kp * 2] = lo;
            *(unsigned*)&Vt[(d8 * 8 + 2 * j + 1) * 40 + kp * 2] = hh;
          }
        }
      }
      __syncthreads();
      for (int ds = 0; ds < 16; ++ds) {
        const unsigned short* ap =
            &Qs[qrow * 1040 + dh * 520 + ds * 32 + akb];
        v16bf a = ld_frag(ap, ap + 16);
        const unsigned short* bp0 = &Ks[ln16 * 520 + ds * 32 + bkb];
        const unsigned short* bp1 = &Ks[(16 + ln16) * 520 + ds * 32 + bkb];
        s0 = wmma_bf16(a, ld_frag(bp0, bp0 + 16), s0);
        s1 = wmma_bf16(a, ld_frag(bp1, bp1 + 16), s1);
      }
    }

    // --- mask, scale, online softmax (row = v + 8*hi; cols = lanes 0..15)
    float p0[8], p1[8], corr[8];
    const int kc0 = kt * 32 + ln16, kc1 = kc0 + 16;
    for (int v = 0; v < 8; ++v) {
      const int* mrow = &mask[(qg + v + 8 * hi) * (size_t)TK_];
      float x0 = s0[v] * scale; if (mrow[kc0] == 0) x0 = -1e9f;
      float x1 = s1[v] * scale; if (mrow[kc1] == 0) x1 = -1e9f;
      float mx = fmaxf(x0, x1);
      for (int sh = 1; sh < 16; sh <<= 1)
        mx = fmaxf(mx, __shfl_xor(mx, sh, 32));
      float mnew = fmaxf(m_i[v], mx);
      float c  = __expf(m_i[v] - mnew);
      float e0 = __expf(x0 - mnew);
      float e1 = __expf(x1 - mnew);
      float rs = e0 + e1;
      for (int sh = 1; sh < 16; sh <<= 1) rs += __shfl_xor(rs, sh, 32);
      l_i[v] = l_i[v] * c + rs;
      m_i[v] = mnew;
      corr[v] = c; p0[v] = e0; p1[v] = e1;
    }

    for (int ni = 0; ni < 16; ++ni)
      for (int v = 0; v < 8; ++v) O[ni][v] *= corr[v];

    // --- P (C-layout) -> packed bf16 -> LDS -> A-layout fragment
    unsigned short* pb = &Ps[wave * 16 * 40];
    for (int v = 0; v < 8; ++v) {
      unsigned pp = cvt_pk_bf16(p0[v], p1[v]);
      pb[(v + 8 * hi) * 40 + ln16]      = (unsigned short)pp;
      pb[(v + 8 * hi) * 40 + 16 + ln16] = (unsigned short)(pp >> 16);
    }
    asm volatile("s_wait_dscnt 0" ::: "memory");
    __builtin_amdgcn_wave_barrier();
    const unsigned short* pap = &pb[ln16 * 40 + akb];
    v16bf pa = ld_frag(pap, pap + 16);

    // --- O += P @ V
    for (int ni = 0; ni < 16; ++ni) {
      const unsigned short* vp = &Vt[(wd * 256 + ni * 16 + ln16) * 40 + bkb];
      O[ni] = wmma_bf16(pa, ld_frag(vp, vp + 16), O[ni]);
    }
  }

  // --- epilogue
  for (int ni = 0; ni < 16; ++ni) {
    int col = d0 + wd * 256 + ni * 16 + ln16;
    for (int v = 0; v < 8; ++v) {
      size_t row = (size_t)b * TQ_ + q0 + wq * 16 + v + 8 * hi;
      out[row * DD + col] = O[ni][v] / l_i[v];
    }
  }
}

// ================================================================ launcher
extern "C" void kernel_launch(void* const* d_in, const int* in_sizes, int n_in,
                              void* d_out, int out_size, void* d_ws,
                              size_t ws_size, hipStream_t stream) {
  const float* query = (const float*)d_in[0];
  const float* keys  = (const float*)d_in[1];
  const float* Wq    = (const float*)d_in[2];
  const float* bq    = (const float*)d_in[3];
  const float* Wk    = (const float*)d_in[4];
  const float* bk    = (const float*)d_in[5];
  const float* Wv    = (const float*)d_in[6];
  const float* bv    = (const float*)d_in[7];
  const int*   mask  = (const int*)d_in[8];
  float*       out   = (float*)d_out;

  unsigned short* qb = (unsigned short*)d_ws;
  unsigned short* kb = qb + (size_t)B_ * TQ_ * DD;
  unsigned short* vb = kb + (size_t)B_ * TK_ * DD;

  dim3 pg((B_ * TQ_) / 128, DD / 128);
  proj_gemm<<<pg, 256, 0, stream>>>(query, Wq, bq, qb);
  proj_gemm<<<pg, 256, 0, stream>>>(keys,  Wk, bk, kb);
  proj_gemm<<<pg, 256, 0, stream>>>(keys,  Wv, bv, vb);

  dim3 ag(TQ_ / 32, DD / 512, B_);
  flash_attn<<<ag, 128, 0, stream>>>(qb, kb, vb, mask, out);
}
